// bVAE_21930103013767
// MI455X (gfx1250) — matrix operand order repaired and checked by reading, not compiled, not run on hardware
//
#include <hip/hip_runtime.h>

// ---------------- problem constants ----------------
#define IN_DIM   512
#define LATENT   64
#define OUT_DIM  512
#define BATCH    131072
#define WAVES    8      // waves per workgroup
#define TPB      256    // 8 waves * 32 lanes (wave32)
#define ROWS_PER_WAVE 16

// ---------------- vector types ----------------
typedef _Float16 v16h __attribute__((ext_vector_type(16)));
typedef _Float16 h8t  __attribute__((ext_vector_type(8)));
typedef float    v8f  __attribute__((ext_vector_type(8)));
typedef float    f4v  __attribute__((ext_vector_type(4)));
typedef int      v4i  __attribute__((ext_vector_type(4)));

// address-space pointer types for the async-LDS builtins:
// signature (from clang diagnostic): (v4i AS1*, v4i AS3*, imm offset, imm cpol)
typedef __attribute__((address_space(1))) v4i* gv4i_t;
typedef __attribute__((address_space(3))) v4i* lv4i_t;

#if __has_builtin(__builtin_amdgcn_global_load_async_to_lds_b128)
#define HAVE_ASYNC_LDS 1
#else
#define HAVE_ASYNC_LDS 0
#endif

#if __has_builtin(__builtin_amdgcn_s_wait_asynccnt)
#define ASYNC_WAIT(n) __builtin_amdgcn_s_wait_asynccnt(n)
#else
#define ASYNC_WAIT(n) asm volatile("s_wait_asynccnt %0" ::"i"(n) : "memory")
#endif

// ---------------- LDS layout (bytes) ----------------
// [0, 204800)        : 8 transposed f16 weights Wt[dout][stride] (stride padded, mult of 8)
// [204800, 217088)   : params f32: b[1024] | g[1024] | beta[1024]
// [217088, 235520)   : per-wave activation staging buffers, 16 x 72 f16 each
// [235520, 272384)   : per-wave x-tile double buffers, 2 x 16 x 36 f32 each
#define LDS_P_OFF   204800
#define LDS_A_OFF   217088
#define LDS_X_OFF   235520
#define LDS_TOTAL   272384
#define ARS  72     // activation LDS row stride in f16 (mult of 8 -> 16B aligned)
#define XRS  36     // x-tile LDS row stride in f32 (36*4 = 144B, 16B aligned)

struct Params {
    const float* x;
    const float* noise;
    const float* W[8];
    const float* b[8];
    const float* g[8];
    const float* beta[8];
    float*       out;
};

// ---------------- helpers ----------------
__device__ __forceinline__ v8f wmma16(v16h a, v16h b, v8f c) {
    // D = A(16x32 f16) * B(32x16 f16) + C(16x16 f32)
    return __builtin_amdgcn_wmma_f32_16x16x32_f16(
        false, a, false, b, (short)0, c, false, false);
}

// A operand from LDS row-major f16 [16][ARS]: lane holds row m=lane&15,
// K chunks [32kt+8hi .. +7] and [32kt+16+8hi .. +7]  (ISA 16-bit A layout)
__device__ __forceinline__ v16h load_a_lds(const _Float16* ab, int m, int hi, int kt) {
    union { v16h v; h8t h[2]; } u;
    u.h[0] = *(const h8t*)(ab + m * ARS + kt * 32 + hi * 8);
    u.h[1] = *(const h8t*)(ab + m * ARS + kt * 32 + 16 + hi * 8);
    return u.v;
}

// B operand from LDS Wt[n][k] f16 (row stride rs): lane holds column n,
// K chunks [32kt+8hi .. +7] and [32kt+16+8hi .. +7]  (same order as A)
__device__ __forceinline__ v16h load_b_lds(const _Float16* wt, int n, int hi, int kt, int rs) {
    const _Float16* base = wt + n * rs + kt * 32 + hi * 8;
    union { v16h v; h8t h[2]; } u;
    u.h[0] = *(const h8t*)(base);
    u.h[1] = *(const h8t*)(base + 16);
    return u.v;
}

// build A operand from 4 f32 quads (chunk0 = K[koff..koff+7], chunk1 = K[koff+16..+23])
__device__ __forceinline__ v16h pack_a_f32(f4v c0, f4v c1, f4v c2, f4v c3) {
    union { v16h v; _Float16 e[16]; } u;
#pragma unroll
    for (int i = 0; i < 4; ++i) {
        u.e[i]      = (_Float16)c0[i];
        u.e[4 + i]  = (_Float16)c1[i];
        u.e[8 + i]  = (_Float16)c2[i];
        u.e[12 + i] = (_Float16)c3[i];
    }
    return u.v;
}

// A operand straight from global f32 row (fallback path)
__device__ __forceinline__ v16h load_a_x(const float* xrow, int kt, int hi) {
    const f4v* p0 = (const f4v*)(xrow + kt * 32 + hi * 8);
    const f4v* p1 = (const f4v*)(xrow + kt * 32 + 16 + hi * 8);
    return pack_a_f32(p0[0], p0[1], p1[0], p1[1]);
}

// A operand from a staged 16x32 f32 x-tile in LDS
__device__ __forceinline__ v16h load_a_xlds(const float* xb, int m, int hi) {
    const f4v* p0 = (const f4v*)(xb + m * XRS + hi * 8);
    const f4v* p1 = (const f4v*)(xb + m * XRS + 16 + hi * 8);
    return pack_a_f32(p0[0], p0[1], p1[0], p1[1]);
}

__device__ __forceinline__ float selu_f(float x) {
    const float a = 1.6732632423543772f, s = 1.0507009873554805f;
    return s * (x > 0.0f ? x : a * expm1f(x));
}

__device__ __forceinline__ float softplus_f(float x) {
    return fmaxf(x, 0.0f) + log1pf(__expf(-fabsf(x)));   // stable log1p(exp(x))
}

// reduce 8 per-row partials across the 16 lanes of each half-wave
__device__ __forceinline__ void half_reduce8(float r[8]) {
#pragma unroll
    for (int mask = 1; mask < 16; mask <<= 1) {
#pragma unroll
        for (int v = 0; v < 8; ++v) r[v] += __shfl_xor(r[v], mask, 32);
    }
}

// GEMM with K=64 (A held in regs as 2 ktiles), NT output 16-col tiles, + bias
template <int NT>
__device__ __forceinline__ void gemm64(const v16h a[2], const _Float16* wt,
                                       const float* bb, int col, int hi, v8f* acc) {
#pragma unroll
    for (int nt = 0; nt < NT; ++nt) {
        v8f c = {};
        c = wmma16(a[0], load_b_lds(wt, nt * 16 + col, hi, 0, 72), c);
        c = wmma16(a[1], load_b_lds(wt, nt * 16 + col, hi, 1, 72), c);
        float bias = bb[nt * 16 + col];
#pragma unroll
        for (int v = 0; v < 8; ++v) c[v] += bias;
        acc[nt] = c;
    }
}

// LayerNorm (+optional SELU) in WMMA C layout; feature dim = 16*NT
template <int NT, bool DO_SELU>
__device__ __forceinline__ void ln_act(v8f* acc, const float* g, const float* bt, int col) {
    float rs[8], rq[8];
#pragma unroll
    for (int v = 0; v < 8; ++v) { rs[v] = 0.0f; rq[v] = 0.0f; }
#pragma unroll
    for (int nt = 0; nt < NT; ++nt)
#pragma unroll
        for (int v = 0; v < 8; ++v) { float y = acc[nt][v]; rs[v] += y; rq[v] += y * y; }
    half_reduce8(rs);
    half_reduce8(rq);
    const float inv = 1.0f / (16.0f * NT);
    float mean[8], rstd[8];
#pragma unroll
    for (int v = 0; v < 8; ++v) {
        mean[v] = rs[v] * inv;
        float var = rq[v] * inv - mean[v] * mean[v];
        rstd[v] = rsqrtf(var + 1e-5f);
    }
#pragma unroll
    for (int nt = 0; nt < NT; ++nt) {
        float gg = g[nt * 16 + col], bb = bt[nt * 16 + col];
#pragma unroll
        for (int v = 0; v < 8; ++v) {
            float y = (acc[nt][v] - mean[v]) * rstd[v] * gg + bb;
            if (DO_SELU) y = selu_f(y);
            acc[nt][v] = y;
        }
    }
}

// write 4 C tiles (16x64) to per-wave LDS row-major f16 (C->A relayout)
__device__ __forceinline__ void store_c_to_a(const v8f* acc, _Float16* ab, int col, int hi) {
#pragma unroll
    for (int nt = 0; nt < 4; ++nt)
#pragma unroll
        for (int v = 0; v < 8; ++v)
            ab[(v + 8 * hi) * ARS + nt * 16 + col] = (_Float16)acc[nt][v];
}

// ---------------- fused VAE kernel ----------------
__global__ void __launch_bounds__(TPB, 1)
vae_fused_kernel(Params p) {
    extern __shared__ char smem[];
    _Float16* WT = (_Float16*)smem;
    float*    PP = (float*)(smem + LDS_P_OFF);

    const int tid = threadIdx.x;

    // layer tables (compile-time; l-loops fully unrolled)
    constexpr int din[8]  = {512, 64, 64, 64, 64, 64, 64, 64};
    constexpr int dout[8] = {64, 64, 64, 128, 64, 64, 64, 512};
    constexpr int wof[8]  = {0, 33280, 37888, 42496, 51712, 56320, 60928, 65536};
    constexpr int wst[8]  = {520, 72, 72, 72, 72, 72, 72, 72};
    constexpr int pof[8]  = {0, 64, 128, 192, 320, 384, 448, 512};

    // ---- stage weights (transposed f16) and params into LDS ----
#pragma unroll
    for (int l = 0; l < 8; ++l) {
        const float* W = p.W[l];
        const int n = din[l] * dout[l];
        for (int i = tid; i < n; i += TPB) {           // i = k*dout + nn (coalesced read)
            int k  = i / dout[l];
            int nn = i - k * dout[l];
            WT[wof[l] + nn * wst[l] + k] = (_Float16)W[i];
        }
    }
#pragma unroll
    for (int l = 0; l < 8; ++l) {
        for (int i = tid; i < dout[l]; i += TPB) {
            PP[pof[l] + i]        = p.b[l][i];
            PP[1024 + pof[l] + i] = p.g[l][i];
            PP[2048 + pof[l] + i] = p.beta[l][i];
        }
    }
    __syncthreads();

    // ---- per-wave setup ----
    const int lane = tid & 31;
    const int wave = tid >> 5;
    const int col  = lane & 15;   // column of C tile / row of A tile
    const int hi   = lane >> 4;   // half-wave select
    const long rowbase = ((long)blockIdx.x * WAVES + wave) * ROWS_PER_WAVE;

    _Float16* ab = (_Float16*)(smem + LDS_A_OFF) + wave * (ROWS_PER_WAVE * ARS);

    // ================= encoder layer 0: 512 -> 64 =================
    v8f acc[4] = {};
#if HAVE_ASYNC_LDS
    {
        // double-buffered async global->LDS copy of 16x32 f32 x-tiles (ASYNCcnt)
        float* xbuf = (float*)(smem + LDS_X_OFF) + wave * (2 * 16 * XRS);
        float* bufs[2] = {xbuf, xbuf + 16 * XRS};

        auto issue_tile = [&](int kt, float* xb) {
            // 2 KB tile = 128 b128 chunks; 4 per lane; consecutive lanes ->
            // consecutive 16B chunks of a row (coalesced 128B runs)
#pragma unroll
            for (int j = 0; j < 4; ++j) {
                int id  = j * 32 + lane;
                int row = id >> 3;
                int ch  = id & 7;
                const float* gsrc = p.x + (rowbase + row) * IN_DIM + kt * 32 + ch * 4;
                float*       ldst = xb + row * XRS + ch * 4;
                __builtin_amdgcn_global_load_async_to_lds_b128(
                    (gv4i_t)(v4i*)gsrc, (lv4i_t)(v4i*)ldst, 0, 0);
            }
        };

        issue_tile(0, bufs[0]);
#pragma unroll
        for (int kt = 0; kt < 16; ++kt) {
            if (kt + 1 < 16) {
                issue_tile(kt + 1, bufs[(kt + 1) & 1]);
                ASYNC_WAIT(4);              // current tile landed; next in flight
            } else {
                ASYNC_WAIT(0);
            }
            asm volatile("" ::: "memory");  // keep LDS reads after the wait
            v16h a1 = load_a_xlds(bufs[kt & 1], col, hi);
#pragma unroll
            for (int nt = 0; nt < 4; ++nt)
                acc[nt] = wmma16(a1, load_b_lds(WT + wof[0], nt * 16 + col, hi, kt, 520), acc[nt]);
        }
    }
#else
    {
        const float* xrow = p.x + (rowbase + col) * IN_DIM;
        __builtin_prefetch(xrow, 0, 3);
#pragma unroll
        for (int kt = 0; kt < 16; ++kt) {
            if (kt + 2 < 16) __builtin_prefetch(xrow + (kt + 2) * 32, 0, 3);
            v16h a1 = load_a_x(xrow, kt, hi);
#pragma unroll
            for (int nt = 0; nt < 4; ++nt)
                acc[nt] = wmma16(a1, load_b_lds(WT + wof[0], nt * 16 + col, hi, kt, 520), acc[nt]);
        }
    }
#endif
#pragma unroll
    for (int nt = 0; nt < 4; ++nt) {
        float bias = PP[pof[0] + nt * 16 + col];
#pragma unroll
        for (int v = 0; v < 8; ++v) acc[nt][v] += bias;
    }
    ln_act<4, true>(acc, PP + 1024 + pof[0], PP + 2048 + pof[0], col);
    store_c_to_a(acc, ab, col, hi);
    v16h a[2];
    a[0] = load_a_lds(ab, col, hi, 0);
    a[1] = load_a_lds(ab, col, hi, 1);

    // ================= encoder layers 1,2: 64 -> 64 =================
#pragma unroll
    for (int l = 1; l <= 2; ++l) {
        gemm64<4>(a, WT + wof[l], PP + pof[l], col, hi, acc);
        ln_act<4, true>(acc, PP + 1024 + pof[l], PP + 2048 + pof[l], col);
        store_c_to_a(acc, ab, col, hi);
        a[0] = load_a_lds(ab, col, hi, 0);
        a[1] = load_a_lds(ab, col, hi, 1);
    }

    // ================= encoder layer 3: 64 -> 128 (mu | logvar) =================
    {
        v8f acc8[8];
        gemm64<8>(a, WT + wof[3], PP + pof[3], col, hi, acc8);
        ln_act<8, true>(acc8, PP + 1024 + pof[3], PP + 2048 + pof[3], col);

        // reparameterize: z = mu + (softplus(logvar)+1e-8) * noise
        v8f z[4];
#pragma unroll
        for (int t = 0; t < 4; ++t) {
#pragma unroll
            for (int v = 0; v < 8; ++v) {
                float nv = p.noise[(rowbase + v + 8 * hi) * LATENT + t * 16 + col];
                float sc = softplus_f(acc8[4 + t][v]) + 1e-8f;
                z[t][v] = acc8[t][v] + sc * nv;
            }
        }
        store_c_to_a(z, ab, col, hi);
        a[0] = load_a_lds(ab, col, hi, 0);
        a[1] = load_a_lds(ab, col, hi, 1);
    }

    // ================= decoder layers 0..2: 64 -> 64 (LN then SELU) =================
#pragma unroll
    for (int l = 4; l <= 6; ++l) {
        gemm64<4>(a, WT + wof[l], PP + pof[l], col, hi, acc);
        ln_act<4, true>(acc, PP + 1024 + pof[l], PP + 2048 + pof[l], col);
        store_c_to_a(acc, ab, col, hi);
        a[0] = load_a_lds(ab, col, hi, 0);
        a[1] = load_a_lds(ab, col, hi, 1);
    }

    // ================= decoder layer 3: 64 -> 512, LN (no SELU), store =================
    {
        const _Float16* wt = WT + wof[7];
        const float* bb = PP + pof[7];
        const float* gg = PP + 1024 + pof[7];
        const float* bt = PP + 2048 + pof[7];

        // pass 1: statistics over 512 features (register-chunked, no spill)
        float rs[8], rq[8];
#pragma unroll
        for (int v = 0; v < 8; ++v) { rs[v] = 0.0f; rq[v] = 0.0f; }
        for (int nt = 0; nt < 32; ++nt) {
            v8f c = {};
            c = wmma16(a[0], load_b_lds(wt, nt * 16 + col, hi, 0, 72), c);
            c = wmma16(a[1], load_b_lds(wt, nt * 16 + col, hi, 1, 72), c);
            float bias = bb[nt * 16 + col];
#pragma unroll
            for (int v = 0; v < 8; ++v) {
                float y = c[v] + bias;
                rs[v] += y;
                rq[v] += y * y;
            }
        }
        half_reduce8(rs);
        half_reduce8(rq);
        float mean[8], rstd[8];
        const float inv = 1.0f / 512.0f;
#pragma unroll
        for (int v = 0; v < 8; ++v) {
            mean[v] = rs[v] * inv;
            float var = rq[v] * inv - mean[v] * mean[v];
            rstd[v] = rsqrtf(var + 1e-5f);
        }

        // pass 2: recompute (GEMM is cheap vs spilling), normalize, store
        for (int nt = 0; nt < 32; ++nt) {
            v8f c = {};
            c = wmma16(a[0], load_b_lds(wt, nt * 16 + col, hi, 0, 72), c);
            c = wmma16(a[1], load_b_lds(wt, nt * 16 + col, hi, 1, 72), c);
            float bias = bb[nt * 16 + col];
            float gv = gg[nt * 16 + col];
            float bv = bt[nt * 16 + col];
#pragma unroll
            for (int v = 0; v < 8; ++v) {
                float y = (c[v] + bias - mean[v]) * rstd[v] * gv + bv;
                p.out[(rowbase + v + 8 * hi) * OUT_DIM + nt * 16 + col] = y;
            }
        }
    }
}

// ---------------- launch ----------------
extern "C" void kernel_launch(void* const* d_in, const int* in_sizes, int n_in,
                              void* d_out, int out_size, void* d_ws, size_t ws_size,
                              hipStream_t stream) {
    (void)in_sizes; (void)n_in; (void)out_size; (void)d_ws; (void)ws_size;
    Params p;
    p.x     = (const float*)d_in[0];
    p.noise = (const float*)d_in[1];
    for (int i = 0; i < 4; ++i) {
        p.W[i]        = (const float*)d_in[2 + i];    // enc_W
        p.b[i]        = (const float*)d_in[6 + i];    // enc_b
        p.g[i]        = (const float*)d_in[10 + i];   // enc_g
        p.beta[i]     = (const float*)d_in[14 + i];   // enc_beta
        p.W[4 + i]    = (const float*)d_in[18 + i];   // dec_W
        p.b[4 + i]    = (const float*)d_in[22 + i];   // dec_b
        p.g[4 + i]    = (const float*)d_in[26 + i];   // dec_g
        p.beta[4 + i] = (const float*)d_in[30 + i];   // dec_beta
    }
    p.out = (float*)d_out;

    (void)hipFuncSetAttribute((const void*)vae_fused_kernel,
                              hipFuncAttributeMaxDynamicSharedMemorySize, LDS_TOTAL);

    const int blocks = BATCH / (WAVES * ROWS_PER_WAVE);  // 1024
    vae_fused_kernel<<<blocks, TPB, LDS_TOTAL, stream>>>(p);
}
